// multiHeadAttention_67491116089519
// MI455X (gfx1250) — matrix operand order repaired and checked by reading, not compiled, not run on hardware
//
#include <hip/hip_runtime.h>

// ---------------------------------------------------------------------------
// MI455X (gfx1250) fused multi-head attention pipeline, bf16 WMMA everywhere.
//   q = x@Wq, k = x@Wk, vT = (x@Wv)^T           (batched bf16 GEMMs, 2x2 wave tiles)
//   axT[b,h,f,i] = (softmax_t(k q^T / E) @ v)^T  (fused flash-style, LDS tiles)
//   out = gather(axT) @ Wlt                      (bf16 GEMM, fp32 out)
// ---------------------------------------------------------------------------

typedef __bf16 bf16;
typedef __attribute__((ext_vector_type(16))) bf16  v16bf;
typedef __attribute__((ext_vector_type(8)))  bf16  bf16x8;
typedef __attribute__((ext_vector_type(4)))  bf16  bf16x4;
typedef __attribute__((ext_vector_type(8)))  float v8f;
typedef __attribute__((ext_vector_type(4)))  float f32x4;

constexpr int BB = 8, SS = 1024, DD = 512, HH = 8, EE = 64;

__device__ __forceinline__ bf16 f2bf(float f) {
  unsigned u = __builtin_bit_cast(unsigned, f);
  u += 0x7FFFu + ((u >> 16) & 1u);                  // round-to-nearest-even
  unsigned short s = (unsigned short)(u >> 16);
  return __builtin_bit_cast(bf16, s);
}

// Build a 16-element bf16 WMMA fragment from two contiguous 8-element chunks.
__device__ __forceinline__ v16bf load_pair(const bf16* p0, const bf16* p1) {
  bf16x8 a = *(const bf16x8*)p0;
  bf16x8 b = *(const bf16x8*)p1;
  v16bf r;
#pragma unroll
  for (int i = 0; i < 8; ++i) { r[i] = a[i]; r[i + 8] = b[i]; }
  return r;
}

__device__ __forceinline__ v8f wmma_bf16(v16bf a, v16bf b, v8f c) {
  return __builtin_amdgcn_wmma_f32_16x16x32_bf16(false, a, false, b,
                                                 (short)0, c, false, false);
}

// ---------------------------------------------------------------------------
__global__ __launch_bounds__(256) void cast_f32_bf16(const float* __restrict__ in,
                                                     bf16* __restrict__ out, long n) {
  long i = ((long)blockIdx.x * 256 + threadIdx.x) * 4;
  if (i + 3 < n) {
    f32x4 v = *(const f32x4*)(in + i);
    bf16x4 o;
    o[0] = f2bf(v[0]); o[1] = f2bf(v[1]); o[2] = f2bf(v[2]); o[3] = f2bf(v[3]);
    *(bf16x4*)(out + i) = o;
  }
}

// ---------------------------------------------------------------------------
// Batched projection GEMM:  O[z] = X[z/H] (MxK, row-major) * W[z%H] (KxN).
// 256 threads = 8 waves arranged WM x WN; each wave owns a 32x32 output block
// (2x2 WMMA tiles) so every staged fragment feeds two WMMAs and each barrier
// pair covers 4 matrix ops. A fragments stream from global (rows contiguous);
// the W k-tile is staged transposed in LDS for contiguous ds_load_b128 reads.
// STORE_T: write O transposed (O[n*M + m]) with packed b128 stores.
// ---------------------------------------------------------------------------
template <int M, int N, int K, int WM, int WN, bool STORE_T>
__global__ __launch_bounds__(256) void gemm_proj(
    const bf16* __restrict__ Xall, long strideXb,
    const bf16* __restrict__ Wall, long strideWh,
    bf16* __restrict__ Oall, long strideOz) {
  constexpr int BM = WM * 32, BN = WN * 32;
  const int z = blockIdx.z;
  const bf16* A = Xall + (long)(z / HH) * strideXb;
  const bf16* W = Wall + (long)(z % HH) * strideWh;
  bf16* O = Oall + (long)z * strideOz;

  const int tid = threadIdx.x;
  const int w = tid >> 5, lane = tid & 31, half = lane >> 4, l16 = lane & 15;
  const int wr = w % WM, wc = w / WM;
  const int m0 = blockIdx.x * BM, n0 = blockIdx.y * BN;
  const int tm = m0 + wr * 32, tn = n0 + wc * 32;

  __shared__ __align__(16) bf16 sBT[BN][40];  // [n_rel][k_rel], 80B rows (16B aligned)

  const int skk = tid >> 3;         // k row staged by this thread
  const int snn = (tid & 7) * 8;    // first n staged by this thread

  v8f acc[2][2] = {};
  for (int kt = 0; kt < K; kt += 32) {
#pragma unroll
    for (int c = 0; c < BN / 64; ++c) {
      bf16x8 wrow = *(const bf16x8*)(W + (long)(kt + skk) * N + n0 + c * 64 + snn);
#pragma unroll
      for (int e = 0; e < 8; ++e) sBT[c * 64 + snn + e][skk] = wrow[e];
    }
    __syncthreads();

    v16bf af[2], bfrag[2];
#pragma unroll
    for (int mi = 0; mi < 2; ++mi) {
      const long arow = tm + mi * 16 + l16;
      af[mi] = load_pair(A + arow * K + kt + half * 8,
                         A + arow * K + kt + 16 + half * 8);
    }
#pragma unroll
    for (int ni = 0; ni < 2; ++ni) {
      const int nrel = wc * 32 + ni * 16 + l16;
      bfrag[ni] = load_pair(&sBT[nrel][half * 8], &sBT[nrel][16 + half * 8]);
    }
#pragma unroll
    for (int mi = 0; mi < 2; ++mi)
#pragma unroll
      for (int ni = 0; ni < 2; ++ni)
        acc[mi][ni] = wmma_bf16(af[mi], bfrag[ni], acc[mi][ni]);
    __syncthreads();
  }

#pragma unroll
  for (int mi = 0; mi < 2; ++mi)
#pragma unroll
    for (int ni = 0; ni < 2; ++ni) {
      if (STORE_T) {
        bf16x8 pk;
#pragma unroll
        for (int j = 0; j < 8; ++j) pk[j] = f2bf(acc[mi][ni][j]);
        *(bf16x8*)(O + (long)(tn + ni * 16 + l16) * M + tm + mi * 16 + half * 8) = pk;
      } else {
#pragma unroll
        for (int j = 0; j < 8; ++j)
          O[(long)(tm + mi * 16 + half * 8 + j) * N + tn + ni * 16 + l16] =
              f2bf(acc[mi][ni][j]);
      }
    }
}

// ---------------------------------------------------------------------------
// Fused attention per (b,h): block owns a 64-row band of i.
// For each 64-wide j tile: S = K·Q^T/64 (WMMA, operands straight from global),
// online softmax over j (running m/l in LDS), then O^T += V^T · P^T so that
// both PV operands are contiguous (A = vT rows from global, B = P rows in LDS).
// Result axT[b,h,f,i] is exactly the layout the final reshape-GEMM wants.
// ---------------------------------------------------------------------------
__global__ __launch_bounds__(256) void attention_fused(
    const bf16* __restrict__ Q, const bf16* __restrict__ Km,
    const bf16* __restrict__ VT, bf16* __restrict__ AxT) {
  const int z = blockIdx.z;
  const bf16* q = Q + (long)z * SS * EE;
  const bf16* k = Km + (long)z * SS * EE;
  const bf16* vt = VT + (long)z * DD * SS;
  bf16* axt = AxT + (long)z * DD * SS;
  const int i0 = blockIdx.x * 64;

  const int tid = threadIdx.x;
  const int w = tid >> 5, lane = tid & 31, half = lane >> 4, l16 = lane & 15;

  __shared__ __align__(16) float sS[64][68];       // raw scores (fp32)
  __shared__ __align__(16) bf16  sP[64][80];       // probabilities (bf16), 160B rows
  __shared__ float sM[64], sL[64], sC[64];         // running max / sum / correction
  __shared__ __align__(16) bf16  sStage[8][16 * 64];  // per-wave output staging

  v8f acc[4][4] = {};   // wave w owns O^T rows f in [w*64, w*64+64), all 64 i cols
  if (tid < 64) { sM[tid] = -3.0e38f; sL[tid] = 0.f; }
  __syncthreads();

  const int fw = w * 64;

  for (int jt = 0; jt < 16; ++jt) {
    const int j0 = jt * 64;

    // (a) score tiles: 16 WMMA tiles, 2 per wave. A=k rows, B=q rows (= Q^T cols).
#pragma unroll
    for (int tt = 0; tt < 2; ++tt) {
      const int t = w * 2 + tt;
      const int it = t >> 2, jj = t & 3;
      const long irow = i0 + it * 16 + l16;
      const long jcol = j0 + jj * 16 + l16;
      v8f cs = {};
#pragma unroll
      for (int ks = 0; ks < EE; ks += 32) {
        v16bf af = load_pair(k + irow * EE + ks + half * 8,
                             k + irow * EE + ks + 16 + half * 8);
        v16bf bfrag = load_pair(q + jcol * EE + ks + half * 8,
                                q + jcol * EE + ks + 16 + half * 8);
        cs = wmma_bf16(af, bfrag, cs);
      }
#pragma unroll
      for (int jv = 0; jv < 8; ++jv)
        sS[it * 16 + half * 8 + jv][jj * 16 + l16] = cs[jv] * (1.0f / 64.0f);
    }
    __syncthreads();

    // (b) online softmax update, one thread per i row.
    if (tid < 64) {
      const int r = tid;
      float mold = sM[r], mx = mold;
#pragma unroll 8
      for (int j = 0; j < 64; ++j) mx = fmaxf(mx, sS[r][j]);
      const float cf = __expf(mold - mx);
      float l = sL[r] * cf;
#pragma unroll 8
      for (int j = 0; j < 64; ++j) {
        const float p = __expf(sS[r][j] - mx);
        l += p;
        sP[r][j] = f2bf(p);
      }
      sM[r] = mx; sL[r] = l; sC[r] = cf;
    }
    __syncthreads();

    // (c) rescale accumulators by exp(m_old - m_new), then O^T += V^T P^T.
    float cf[4];
#pragma unroll
    for (int ii = 0; ii < 4; ++ii) cf[ii] = sC[ii * 16 + l16];
#pragma unroll
    for (int fi = 0; fi < 4; ++fi)
#pragma unroll
      for (int ii = 0; ii < 4; ++ii) acc[fi][ii] *= cf[ii];

#pragma unroll
    for (int ks = 0; ks < 64; ks += 32) {
      v16bf afr[4], bfr[4];
#pragma unroll
      for (int fi = 0; fi < 4; ++fi) {
        const long frow = fw + fi * 16 + l16;
        afr[fi] = load_pair(vt + frow * SS + j0 + ks + half * 8,
                            vt + frow * SS + j0 + ks + 16 + half * 8);
      }
#pragma unroll
      for (int ii = 0; ii < 4; ++ii)
        bfr[ii] = load_pair(&sP[ii * 16 + l16][ks + half * 8],
                            &sP[ii * 16 + l16][ks + 16 + half * 8]);
#pragma unroll
      for (int fi = 0; fi < 4; ++fi)
#pragma unroll
        for (int ii = 0; ii < 4; ++ii)
          acc[fi][ii] = wmma_bf16(afr[fi], bfr[ii], acc[fi][ii]);
    }
    __syncthreads();
  }

  // Finalize: divide by softmax denominator, stage per-wave 16x64 tiles in LDS,
  // then coalesced b32 stores of axT rows (f-major, i-contiguous).
  float rl[4];
#pragma unroll
  for (int ii = 0; ii < 4; ++ii) rl[ii] = 1.0f / sL[ii * 16 + l16];

#pragma unroll
  for (int fi = 0; fi < 4; ++fi) {
#pragma unroll
    for (int ii = 0; ii < 4; ++ii)
#pragma unroll
      for (int jv = 0; jv < 8; ++jv)
        sStage[w][(half * 8 + jv) * 64 + ii * 16 + l16] =
            f2bf(acc[fi][ii][jv] * rl[ii]);
    __syncthreads();
    const unsigned* sw32 = (const unsigned*)&sStage[w][0];
#pragma unroll
    for (int r = 0; r < 16; ++r) {
      const int f = fw + fi * 16 + r;
      *(unsigned*)(axt + (long)f * SS + i0 + lane * 2) = sw32[r * 32 + lane];
    }
    __syncthreads();
  }
}

// ---------------------------------------------------------------------------
// Final GEMM: Out[8192,512] (fp32) = Aflat[8192,4096] @ Wlt[4096,512].
// Aflat row r = 4 contiguous 1024-element runs of axT:
//   b = r>>10, f = (r&1023)>>1, h = (r&1)*4 + (c>>10), i = c&1023.
// Same 2x2-wave-tile structure as gemm_proj (WM=2, WN=4 -> 64x128 macro-tile).
// ---------------------------------------------------------------------------
__global__ __launch_bounds__(256) void final_gemm(
    const bf16* __restrict__ AxT, const bf16* __restrict__ Wl,
    float* __restrict__ Out) {
  const int tid = threadIdx.x;
  const int w = tid >> 5, lane = tid & 31, half = lane >> 4, l16 = lane & 15;
  const int wr = w % 2, wc = w / 2;
  const int m0 = blockIdx.x * 64, n0 = blockIdx.y * 128;
  const int tm = m0 + wr * 32, tn = n0 + wc * 32;

  __shared__ __align__(16) bf16 sBT[128][40];
  const int skk = tid >> 3;
  const int snn = (tid & 7) * 8;

  // Per-M-tile gather geometry (row of the implicit reshaped A matrix).
  long abase[2];
  int hb[2];
#pragma unroll
  for (int mi = 0; mi < 2; ++mi) {
    const int r = tm + mi * 16 + l16;
    const int b = r >> 10;
    const int f = (r & 1023) >> 1;
    hb[mi] = (r & 1) * 4;
    abase[mi] = ((long)(b * HH) * DD + f) * SS;  // + h*DD*SS + i
  }

  v8f acc[2][2] = {};
  for (int kt = 0; kt < 4096; kt += 32) {
#pragma unroll
    for (int c = 0; c < 2; ++c) {
      bf16x8 wrow = *(const bf16x8*)(Wl + (long)(kt + skk) * 512 + n0 + c * 64 + snn);
#pragma unroll
      for (int e = 0; e < 8; ++e) sBT[c * 64 + snn + e][skk] = wrow[e];
    }
    __syncthreads();

    v16bf af[2], bfrag[2];
#pragma unroll
    for (int mi = 0; mi < 2; ++mi) {
      const int h = hb[mi] + (kt >> 10);
      const bf16* arow = AxT + abase[mi] + (long)h * DD * SS + (kt & 1023);
      af[mi] = load_pair(arow + half * 8, arow + 16 + half * 8);
    }
#pragma unroll
    for (int ni = 0; ni < 2; ++ni) {
      const int nrel = wc * 32 + ni * 16 + l16;
      bfrag[ni] = load_pair(&sBT[nrel][half * 8], &sBT[nrel][16 + half * 8]);
    }
#pragma unroll
    for (int mi = 0; mi < 2; ++mi)
#pragma unroll
      for (int ni = 0; ni < 2; ++ni)
        acc[mi][ni] = wmma_bf16(af[mi], bfrag[ni], acc[mi][ni]);
    __syncthreads();
  }

#pragma unroll
  for (int mi = 0; mi < 2; ++mi)
#pragma unroll
    for (int ni = 0; ni < 2; ++ni)
#pragma unroll
      for (int j = 0; j < 8; ++j)   // lanes 0..15 hit consecutive n -> coalesced b32
        Out[(long)(tm + mi * 16 + half * 8 + j) * 512 + tn + ni * 16 + l16] =
            acc[mi][ni][j];
}

// ---------------------------------------------------------------------------
extern "C" void kernel_launch(void* const* d_in, const int* in_sizes, int n_in,
                              void* d_out, int out_size, void* d_ws, size_t ws_size,
                              hipStream_t stream) {
  (void)in_sizes; (void)n_in; (void)out_size; (void)ws_size;
  const float* x   = (const float*)d_in[0];
  const float* Wq  = (const float*)d_in[1];
  const float* Wk  = (const float*)d_in[2];
  const float* Wv  = (const float*)d_in[3];
  const float* Wlt = (const float*)d_in[4];
  float* out = (float*)d_out;

  char* ws = (char*)d_ws;
  auto carve = [&](size_t bytes) -> bf16* {
    char* p = ws;
    ws += (bytes + 255) & ~(size_t)255;
    return (bf16*)p;
  };
  const long nX  = (long)BB * SS * DD;        // 4M
  const long nWq = (long)HH * DD * EE;        // 256K
  const long nWv = (long)HH * DD * DD;        // 2M
  const long nWl = (long)HH * DD * DD;        // 2M (= 4096*512)
  const long nQ  = (long)BB * HH * SS * EE;   // 4M
  const long nV  = (long)BB * HH * DD * SS;   // 32M

  bf16* xb   = carve(nX * 2);
  bf16* wqb  = carve(nWq * 2);
  bf16* wkb  = carve(nWq * 2);
  bf16* wvb  = carve(nWv * 2);
  bf16* wlb  = carve(nWl * 2);
  bf16* qb   = carve(nQ * 2);
  bf16* kb   = carve(nQ * 2);
  bf16* vtb  = carve(nV * 2);
  bf16* axtb = carve(nV * 2);

  auto cast = [&](const float* src, bf16* dst, long n) {
    cast_f32_bf16<<<dim3((unsigned)((n + 1023) / 1024)), 256, 0, stream>>>(src, dst, n);
  };
  cast(x, xb, nX);
  cast(Wq, wqb, nWq);
  cast(Wk, wkb, nWq);
  cast(Wv, wvb, nWv);
  cast(Wlt, wlb, nWl);

  // q,k: [S,D]x[D,E] per (b,h), 128x64 macro-tiles (WM=4, WN=2).
  gemm_proj<SS, EE, DD, 4, 2, false><<<dim3(SS / 128, 1, BB * HH), 256, 0, stream>>>(
      xb, (long)SS * DD, wqb, (long)DD * EE, qb, (long)SS * EE);
  gemm_proj<SS, EE, DD, 4, 2, false><<<dim3(SS / 128, 1, BB * HH), 256, 0, stream>>>(
      xb, (long)SS * DD, wkb, (long)DD * EE, kb, (long)SS * EE);
  // v: [S,D]x[D,D], stored transposed as vT[f][i]; 64x128 macro-tiles (WM=2, WN=4).
  gemm_proj<SS, DD, DD, 2, 4, true><<<dim3(SS / 64, DD / 128, BB * HH), 256, 0, stream>>>(
      xb, (long)SS * DD, wvb, (long)DD * DD, vtb, (long)DD * SS);

  attention_fused<<<dim3(SS / 64, 1, BB * HH), 256, 0, stream>>>(qb, kb, vtb, axtb);

  final_gemm<<<dim3(8192 / 64, 512 / 128, 1), 256, 0, stream>>>(axtb, wlb, out);
}
